// FragmentEmbedder_45157286150941
// MI455X (gfx1250) — matrix operand order repaired and checked by reading, not compiled, not run on hardware
//
#include <hip/hip_runtime.h>
#include <math.h>

typedef __attribute__((ext_vector_type(16))) _Float16 v16h;
typedef __attribute__((ext_vector_type(8)))  _Float16 v8h;
typedef __attribute__((ext_vector_type(8)))  float    v8f;

#define NFRAG        524288
#define NREG         16
#define NFEAT        40
#define DIM          64
#define ROWS_PER_BLK 128
#define NTHREADS     256

// 0.5*x*(1+tanh(z)) == x * sigmoid(2z) == x / (1 + exp(-2z)); branch-free,
// one v_exp_f32 + fast reciprocal. Saturates correctly at +/-inf.
__device__ __forceinline__ float gelu_fast(float x) {
    float z = 0.79788456080286535588f * (x + 0.044715f * x * x * x);
    return __fdividef(x, 1.0f + __expf(-2.0f * z));
}

__device__ __forceinline__ v16h cat8h(v8h lo, v8h hi) {
    return __builtin_shufflevector(lo, hi, 0, 1, 2, 3, 4, 5, 6, 7,
                                           8, 9, 10, 11, 12, 13, 14, 15);
}

// MODE: 0 = hidden, no GELU (encoder) -> f16 act; 1 = hidden + GELU -> f16 act;
//       2 = final + GELU -> masked f32 global store.
// Compile-time MODE + full unrolling keeps acc[][] statically indexed
// (no v_movrels / M0 indexing in the epilogue).
template <int MODE>
__device__ __forceinline__ void run_layer(const _Float16* __restrict__ W,
                                          const float*    __restrict__ bias,
                                          _Float16*       __restrict__ act,
                                          float*          __restrict__ out,
                                          const int*      __restrict__ rmap,
                                          int rowbase, int r,
                                          int mrow, int halfsel, int hi8) {
    v8f acc[4];
    // Fold bias into C: per-column value, identical across the 8 C VGPRs.
#pragma unroll
    for (int nt = 0; nt < 4; ++nt) {
        float b = bias[nt * 16 + mrow];
        v8f a;
#pragma unroll
        for (int v = 0; v < 8; ++v) a[v] = b;
        acc[nt] = a;
    }

#pragma unroll
    for (int ks = 0; ks < 2; ++ks) {
        // A fragment (16x32 f16): lane half -> K+8; V4..7 -> K+16
        const v8h* pa = (const v8h*)(act + mrow * DIM + ks * 32 + hi8);
        v16h A = cat8h(pa[0], pa[2]);
#pragma unroll
        for (int nt = 0; nt < 4; ++nt) {
            // B fragment (32x16 f16): column n, K-major contiguous
            const v8h* pb =
                (const v8h*)(W + (nt * 16 + mrow) * DIM + ks * 32 + hi8);
            v16h B = cat8h(pb[0], pb[2]);
            acc[nt] = __builtin_amdgcn_wmma_f32_16x16x32_f16(
                false, A, false, B, (short)0, acc[nt], false, false);
        }
    }

    if (MODE < 2) {
#pragma unroll
        for (int nt = 0; nt < 4; ++nt) {
#pragma unroll
            for (int v = 0; v < 8; ++v) {
                float g = acc[nt][v];
                if (MODE == 1) g = gelu_fast(g);
                act[(v + hi8) * DIM + nt * 16 + mrow] = (_Float16)g;
            }
        }
    } else {
        // final: GELU + masked store (honors sorted-region dispatch)
        bool ok[8];
#pragma unroll
        for (int v = 0; v < 8; ++v) ok[v] = (rmap[rowbase + v + hi8] == r);
#pragma unroll
        for (int nt = 0; nt < 4; ++nt) {
#pragma unroll
            for (int v = 0; v < 8; ++v) {
                if (ok[v]) {
                    out[(rowbase + v + hi8) * DIM + nt * 16 + mrow] =
                        gelu_fast(acc[nt][v]);
                }
            }
        }
    }
}

__global__ __launch_bounds__(NTHREADS)
void FragmentEmbedder_kernel(const int*   __restrict__ coords,   // [N,2]
                             const int*   __restrict__ rmap,     // [N] sorted
                             const float* __restrict__ enc_w,    // [16,40,64]
                             const float* __restrict__ enc_b,    // [16,64]
                             const float* __restrict__ lin_w,    // [2,16,64,64]
                             const float* __restrict__ lin_b,    // [2,16,64]
                             float*       __restrict__ out) {    // [N,64]
    // Weights staged transposed: sW[layer][n*64 + k] (f16), k padded to 64.
    __shared__ __align__(16) _Float16 sW[3][DIM * DIM];
    __shared__ float sB[3][DIM];
    // Per-wave activation tile [16 rows][64 cols] f16.
    __shared__ __align__(16) _Float16 sAct[NTHREADS / 32][16 * DIM];

    const int tid      = threadIdx.x;
    const int wave     = tid >> 5;
    const int lane     = tid & 31;
    const int blockRow = blockIdx.x * ROWS_PER_BLK;
    const int rowbase  = blockRow + wave * 16;

    const int mrow    = lane & 15;   // row within 16 (A) / col within 16 (B,C)
    const int halfsel = lane & 16;   // 0 or 16
    const int hi8     = halfsel >> 1;// 0 or 8

    // regionmapping is sorted -> contiguous region range for this block
    const int r0 = rmap[blockRow];
    const int r1 = rmap[blockRow + ROWS_PER_BLK - 1];

    _Float16* act = sAct[wave];

    for (int r = r0; r <= r1; ++r) {
        __syncthreads();  // previous iteration's weight readers are done

        // ---- stage region-r weights (f32 -> f16, transposed [n][k]) ----
        for (int idx = tid; idx < DIM * DIM; idx += NTHREADS) {
            int n = idx >> 6, k = idx & 63;
            sW[0][idx] = (k < NFEAT) ? (_Float16)enc_w[(r * NFEAT + k) * DIM + n]
                                     : (_Float16)0.0f;
            sW[1][idx] = (_Float16)lin_w[((0 * NREG + r) * DIM + k) * DIM + n];
            sW[2][idx] = (_Float16)lin_w[((1 * NREG + r) * DIM + k) * DIM + n];
        }
        if (tid < DIM) {
            sB[0][tid] = enc_b[r * DIM + tid];
            sB[1][tid] = lin_b[(0 * NREG + r) * DIM + tid];
            sB[2][tid] = lin_b[(1 * NREG + r) * DIM + tid];
        }
        __syncthreads();

        // ---- sinusoidal features for this wave's 16 rows (K padded to 64) --
        // angle = pos * 2^f * pi = 2*pi * (coord * 2^f / 400000).
        // Reduce to one revolution first so native v_sin/v_cos stay in range
        // (raw arg reaches 512*pi > hardware's +/-256*pi window).
        for (int idx = lane; idx < 16 * DIM; idx += 32) {
            int m = idx >> 6, j = idx & 63;
            _Float16 v = (_Float16)0.0f;
            if (j < NFEAT) {
                int c  = coords[(rowbase + m) * 2 + (j >= 20 ? 1 : 0)];
                int f  = (j >= 20) ? (j - 20) : j;     // [sin f0..9 | cos f0..9]
                int fr = (f >= 10) ? (f - 10) : f;
                float rev = (float)c * ((float)(1 << fr) * 2.5e-6f); // /400000
                float t   = rev - floorf(rev);                       // [0,1)
                float ang = t * 6.28318530717958647693f;             // [0,2pi)
                v = (_Float16)((f < 10) ? __sinf(ang) : __cosf(ang));
            }
            act[idx] = v;
        }

        // ---- three per-region layers: enc(40->64), lin0+GELU, lin1+GELU ---
        run_layer<0>(sW[0], sB[0], act, out, rmap, rowbase, r, mrow, halfsel, hi8);
        run_layer<1>(sW[1], sB[1], act, out, rmap, rowbase, r, mrow, halfsel, hi8);
        run_layer<2>(sW[2], sB[2], act, out, rmap, rowbase, r, mrow, halfsel, hi8);
    }
}

extern "C" void kernel_launch(void* const* d_in, const int* in_sizes, int n_in,
                              void* d_out, int out_size, void* d_ws, size_t ws_size,
                              hipStream_t stream) {
    (void)in_sizes; (void)n_in; (void)out_size; (void)d_ws; (void)ws_size;
    const int*   coords = (const int*)d_in[0];
    const int*   rmap   = (const int*)d_in[1];
    const float* enc_w  = (const float*)d_in[2];
    const float* enc_b  = (const float*)d_in[3];
    const float* lin_w  = (const float*)d_in[4];
    const float* lin_b  = (const float*)d_in[5];
    float* out = (float*)d_out;

    dim3 grid(NFRAG / ROWS_PER_BLK);   // 4096 blocks x 256 threads (8 waves)
    FragmentEmbedder_kernel<<<grid, NTHREADS, 0, stream>>>(
        coords, rmap, enc_w, enc_b, lin_w, lin_b, out);
}
